// diffConv_10411000726291
// MI455X (gfx1250) — compile-verified, matrix operand order checked by reading, not run on hardware
//
#include <hip/hip_runtime.h>
#include <math.h>

// ---------------------------------------------------------------------------
// diffConv for MI455X (gfx1250): wave32 + bf16 WMMA pipeline.
// B=4, N=4096, C=64, M=2048, OUT=128, HID=16, K_NEIGH=20 (cent_num hardcoded).
// ---------------------------------------------------------------------------

#define Bb   4
#define Nn   4096
#define Cc   64
#define Mm   2048
#define OUTC 128
#define HID  16
#define KNB  20

typedef __attribute__((ext_vector_type(16))) __bf16 bf16x16;
typedef __attribute__((ext_vector_type(8)))  float  floatx8;

__device__ __forceinline__ float geluf(float v) {
    return 0.5f * v * (1.0f + erff(v * 0.70710678118654752440f));
}

__device__ __forceinline__ float wave_sum(float v) {
    #pragma unroll
    for (int off = 16; off; off >>= 1) v += __shfl_xor(v, off, 32);
    return v;
}

__device__ __forceinline__ unsigned pack_bf16x2(float lo, float hi) {
    unsigned short a = __builtin_bit_cast(unsigned short, (__bf16)lo);
    unsigned short b = __builtin_bit_cast(unsigned short, (__bf16)hi);
    return ((unsigned)b << 16) | (unsigned)a;
}

// One loader for every WMMA operand: tiles live in LDS as [row][32 k-contig]
// (A row-major; B/W/x/k tiles stored transposed so their lane pattern becomes
// two contiguous 8x bf16 runs -> 2x ds_load_b128 per operand).
__device__ __forceinline__ bf16x16 lds_load_tile16x32(const __bf16* p, int lane) {
    bf16x16 a;
    int r  = lane & 15;
    int kh = (lane >> 4) << 3;
    #pragma unroll
    for (int e = 0; e < 16; ++e) {
        int k = ((e >> 3) << 4) + kh + (e & 7);
        a[e] = p[r * 32 + k];
    }
    return a;
}

// ---------------------------------------------------------------------------
__global__ void zero_kernel(float* p, int n) {
    int i = blockIdx.x * blockDim.x + threadIdx.x;
    if (i < n) p[i] = 0.0f;
}

__global__ void xyz2_kernel(const float* __restrict__ xyz, float* __restrict__ xyz2) {
    int i = blockIdx.x * blockDim.x + threadIdx.x;   // B*N
    if (i < Bb * Nn) {
        float a = xyz[i * 3 + 0], b = xyz[i * 3 + 1], c = xyz[i * 3 + 2];
        xyz2[i] = a * a + b * b + c * c;
    }
}

// q_raw[B,M,16], k_raw[B,N,16] + per-channel sum/sumsq (stats = qsum,qss,ksum,kss)
__global__ __launch_bounds__(256)
void qk_raw_kernel(const float* __restrict__ x, const float* __restrict__ xyz,
                   const float* __restrict__ Wq, const float* __restrict__ Wk,
                   float* __restrict__ qraw, float* __restrict__ kraw,
                   float* __restrict__ stats) {
    __shared__ float red[64];
    if (threadIdx.x < 64) red[threadIdx.x] = 0.0f;
    __syncthreads();

    int t = blockIdx.x * 256 + threadIdx.x;          // 0..16383
    int b = t >> 12, n = t & (Nn - 1);
    float fin[67];
    #pragma unroll 4
    for (int c = 0; c < Cc; ++c) fin[c] = x[(size_t)t * Cc + c];
    fin[64] = xyz[t * 3 + 0]; fin[65] = xyz[t * 3 + 1]; fin[66] = xyz[t * 3 + 2];

    #pragma unroll
    for (int o = 0; o < HID; ++o) {
        float acc = 0.0f;
        for (int c = 0; c < 67; ++c) acc += fin[c] * Wk[o * 67 + c];
        kraw[(size_t)t * HID + o] = acc;
        atomicAdd(&red[32 + o], acc);
        atomicAdd(&red[48 + o], acc * acc);
    }
    if (n < Mm) {
        int gm = b * Mm + n;
        #pragma unroll
        for (int o = 0; o < HID; ++o) {
            float acc = 0.0f;
            for (int c = 0; c < 67; ++c) acc += fin[c] * Wq[o * 67 + c];
            qraw[(size_t)gm * HID + o] = acc;
            atomicAdd(&red[o], acc);
            atomicAdd(&red[16 + o], acc * acc);
        }
    }
    __syncthreads();
    if (threadIdx.x < 64) atomicAdd(&stats[threadIdx.x], red[threadIdx.x]);
}

__global__ void bn_finalize_kernel(const float* __restrict__ sum, const float* __restrict__ ssq,
                                   const float* __restrict__ gamma, const float* __restrict__ beta,
                                   float* __restrict__ scale, float* __restrict__ shift,
                                   float cnt, int C) {
    int i = blockIdx.x * blockDim.x + threadIdx.x;
    if (i < C) {
        float m = sum[i] / cnt;
        float v = ssq[i] / cnt - m * m;
        float s = gamma[i] / sqrtf(v + 1e-5f);
        scale[i] = s;
        shift[i] = beta[i] - m * s;
    }
}

__global__ void apply_affine_kernel(const float* __restrict__ in, float* __restrict__ out,
                                    const float* __restrict__ scale, const float* __restrict__ shift,
                                    int n, int C, int act) {
    int i = blockIdx.x * blockDim.x + threadIdx.x;
    if (i < n) {
        int c = i % C;
        float v = in[i] * scale[c] + shift[c];
        out[i] = act ? geluf(v) : v;
    }
}

// kd[b,m] = sum_n exp(-dist/0.02); one wave per (b,m)
__global__ __launch_bounds__(32)
void kd_kernel(const float* __restrict__ xyz, const float* __restrict__ xyz2,
               float* __restrict__ kd) {
    int gm = blockIdx.x;                              // b*M+m
    int b = gm >> 11, m = gm & (Mm - 1), lane = threadIdx.x;
    const float* cen = &xyz[((size_t)b * Nn + m) * 3];
    float cx = cen[0], cy = cen[1], cz = cen[2];
    float c2 = xyz2[b * Nn + m];
    float acc = 0.0f;
    for (int n = lane; n < Nn; n += 32) {
        const float* p = &xyz[((size_t)b * Nn + n) * 3];
        __builtin_prefetch(&xyz[((size_t)b * Nn + n + 256) * 3], 0, 1);
        float d = c2 + xyz2[b * Nn + n] - 2.0f * (cx * p[0] + cy * p[1] + cz * p[2]);
        acc += expf(-d * 50.0f);
    }
    acc = wave_sum(acc);
    if (lane == 0) kd[gm] = acc;
}

__global__ __launch_bounds__(256)
void radius_kernel(const float* __restrict__ kd, float* __restrict__ radius) {
    int b = blockIdx.x, t = threadIdx.x;
    __shared__ float red[256];
    float mx = -1e30f;
    for (int m = t; m < Mm; m += 256) mx = fmaxf(mx, kd[b * Mm + m]);
    red[t] = mx;
    __syncthreads();
    for (int off = 128; off; off >>= 1) {
        if (t < off) red[t] = fmaxf(red[t], red[t + off]);
        __syncthreads();
    }
    float kmax = red[0];
    for (int m = t; m < Mm; m += 256) {
        float sc = kd[b * Mm + m] / (kmax + 1e-9f);
        radius[b * Mm + m] = 0.1f + 0.1f * sc;
    }
}

// online max/sum of masked logits, one wave per (b,m)
__global__ __launch_bounds__(32)
void softmax_stats_kernel(const float* __restrict__ xyz, const float* __restrict__ xyz2,
                          const float* __restrict__ qn, const float* __restrict__ kn,
                          const float* __restrict__ radius,
                          float* __restrict__ rowmax, float* __restrict__ rowsum) {
    int gm = blockIdx.x;
    int b = gm >> 11, m = gm & (Mm - 1), lane = threadIdx.x;
    float qv[HID];
    #pragma unroll
    for (int h = 0; h < HID; ++h) qv[h] = qn[(size_t)gm * HID + h];
    const float* cen = &xyz[((size_t)b * Nn + m) * 3];
    float cx = cen[0], cy = cen[1], cz = cen[2];
    float c2 = xyz2[b * Nn + m], rad = radius[gm];
    float mx = -3.0e38f, sum = 0.0f;
    for (int n = lane; n < Nn; n += 32) {
        const float* kk = &kn[((size_t)b * Nn + n) * HID];
        float lg = 0.0f;
        #pragma unroll
        for (int h = 0; h < HID; ++h) lg += qv[h] * kk[h];
        const float* p = &xyz[((size_t)b * Nn + n) * 3];
        float d = c2 + xyz2[b * Nn + n] - 2.0f * (cx * p[0] + cy * p[1] + cz * p[2]);
        float l = (d < rad) ? lg : -1.0e9f;
        if (l > mx) { sum = sum * expf(mx - l); mx = l; }
        sum += expf(l - mx);
    }
    #pragma unroll
    for (int off = 16; off; off >>= 1) {
        float omx = __shfl_xor(mx, off, 32);
        float osm = __shfl_xor(sum, off, 32);
        float nm = fmaxf(mx, omx);
        sum = sum * expf(mx - nm) + osm * expf(omx - nm);
        mx = nm;
    }
    if (lane == 0) { rowmax[gm] = mx; rowsum[gm] = sum; }
}

// s1[b,n] = sum_m a[b,m,n]; one wave per (b,n) column
__global__ __launch_bounds__(32)
void colsum_kernel(const float* __restrict__ xyz, const float* __restrict__ xyz2,
                   const float* __restrict__ qn, const float* __restrict__ kn,
                   const float* __restrict__ radius, const float* __restrict__ rowmax,
                   const float* __restrict__ rowsum, float* __restrict__ s1) {
    int gn = blockIdx.x;                               // b*N+n
    int b = gn >> 12, lane = threadIdx.x;
    float kv[HID];
    #pragma unroll
    for (int h = 0; h < HID; ++h) kv[h] = kn[(size_t)gn * HID + h];
    const float* p = &xyz[(size_t)gn * 3];
    float px = p[0], py = p[1], pz = p[2], x2 = xyz2[gn];
    float acc = 0.0f;
    for (int m = lane; m < Mm; m += 32) {
        int gm = b * Mm + m;
        const float* q = &qn[(size_t)gm * HID];
        float lg = 0.0f;
        #pragma unroll
        for (int h = 0; h < HID; ++h) lg += q[h] * kv[h];
        const float* c = &xyz[((size_t)b * Nn + m) * 3];
        float d = xyz2[b * Nn + m] + x2 - 2.0f * (c[0] * px + c[1] * py + c[2] * pz);
        float maskf = (d < radius[gm]) ? 1.0f : 0.0f;
        float l = (maskf > 0.0f) ? lg : -1.0e9f;
        float pr = expf(l - rowmax[gm]) / rowsum[gm];
        acc += sqrtf(maskf + 1e-9f) * sqrtf(pr + 1e-9f) - 1e-9f;   // a >= 0
    }
    acc = wave_sum(acc);
    if (lane == 0) s1[gn] = acc;
}

// smoothed = adj'' @ x, with logits AND the smoothing GEMM on the matrix core.
// Per 32-n chunk: 2 WMMAs compute the 16x32 logit tile (q zero-padded to K=32),
// the masked-softmax/normalize epilogue runs in the WMMA D layout, and 4 WMMAs
// accumulate smoothed over the 64 feature channels. Fuses the s2 row-norm and
// writes the 128-wide feat-GEMM input [smoothed - cent_feat | cent_feat].
__global__ __launch_bounds__(32)
void smoothed_wmma_kernel(const float* __restrict__ x, const float* __restrict__ xyz,
                          const float* __restrict__ xyz2, const float* __restrict__ qn,
                          const float* __restrict__ kn, const float* __restrict__ radius,
                          const float* __restrict__ rowmax, const float* __restrict__ rowsum,
                          const float* __restrict__ s1, float* __restrict__ in_feat) {
    int b = blockIdx.y;
    int mt = blockIdx.x * 16;
    int lane = threadIdx.x;
    int nl = lane & 15, hi = lane >> 4;

    __shared__ float sC[16][8];                        // cx,cy,cz,c2,rad,rmax,rsum
    __shared__ __align__(16) __bf16 sQ[16][32];        // q tile, k-padded
    __shared__ __align__(16) __bf16 sKt[32][32];       // k chunk, n-major (transposed)
    __shared__ __align__(16) __bf16 sA[16][32];        // adj tile (m x n)
    __shared__ __align__(16) __bf16 sXt[64][32];       // x chunk, c-major (transposed)

    for (int t = lane; t < 512; t += 32) {             // stage q (zero-pad K 16->32)
        int m = t >> 5, k = t & 31;
        float v = (k < HID) ? qn[((size_t)(b * Mm + mt + m)) * HID + k] : 0.0f;
        sQ[m][k] = (__bf16)v;
    }
    if (lane < 16) {
        int gm = b * Mm + mt + lane;
        const float* cen = &xyz[((size_t)b * Nn + mt + lane) * 3];
        sC[lane][0] = cen[0]; sC[lane][1] = cen[1]; sC[lane][2] = cen[2];
        sC[lane][3] = xyz2[b * Nn + mt + lane];
        sC[lane][4] = radius[gm];
        sC[lane][5] = rowmax[gm];
        sC[lane][6] = rowsum[gm];
    }
    __syncthreads();
    bf16x16 aQ = lds_load_tile16x32(&sQ[0][0], lane);

    floatx8 acc[4] = {};
    float sacc8[8];
    #pragma unroll
    for (int v = 0; v < 8; ++v) sacc8[v] = 0.0f;

    for (int nc = 0; nc < Nn; nc += 32) {
        // --- stage k chunk transposed: lane = n row, pack 16 bf16 + 16 zeros
        {
            const float4* kr = (const float4*)&kn[((size_t)b * Nn + nc + lane) * HID];
            float4 k0 = kr[0], k1 = kr[1], k2 = kr[2], k3 = kr[3];
            *(uint4*)&sKt[lane][0] = make_uint4(pack_bf16x2(k0.x, k0.y), pack_bf16x2(k0.z, k0.w),
                                                pack_bf16x2(k1.x, k1.y), pack_bf16x2(k1.z, k1.w));
            *(uint4*)&sKt[lane][8] = make_uint4(pack_bf16x2(k2.x, k2.y), pack_bf16x2(k2.z, k2.w),
                                                pack_bf16x2(k3.x, k3.y), pack_bf16x2(k3.z, k3.w));
            *(uint4*)&sKt[lane][16] = make_uint4(0u, 0u, 0u, 0u);
            *(uint4*)&sKt[lane][24] = make_uint4(0u, 0u, 0u, 0u);
        }
        // --- stage x chunk transposed (c-major): coalesced dword loads over c
        #pragma unroll
        for (int half = 0; half < 2; ++half) {
            int c = lane + 32 * half;
            #pragma unroll
            for (int kg = 0; kg < 4; ++kg) {
                float f[8];
                #pragma unroll
                for (int j = 0; j < 8; ++j)
                    f[j] = x[((size_t)b * Nn + nc + kg * 8 + j) * Cc + c];
                *(uint4*)&sXt[c][kg * 8] =
                    make_uint4(pack_bf16x2(f[0], f[1]), pack_bf16x2(f[2], f[3]),
                               pack_bf16x2(f[4], f[5]), pack_bf16x2(f[6], f[7]));
            }
        }
        __syncthreads();

        // --- logits via WMMA: two 16x16 subtiles of q @ k^T
        floatx8 lg0, lg1;
        {
            floatx8 z = {};
            bf16x16 bK0 = lds_load_tile16x32(&sKt[0][0], lane);
            lg0 = __builtin_amdgcn_wmma_f32_16x16x32_bf16(false, aQ, false, bK0,
                                                          (short)0, z, false, false);
            bf16x16 bK1 = lds_load_tile16x32(&sKt[16][0], lane);
            lg1 = __builtin_amdgcn_wmma_f32_16x16x32_bf16(false, aQ, false, bK1,
                                                          (short)0, z, false, false);
        }
        // --- masked-softmax / double-norm epilogue in the D layout
        #pragma unroll
        for (int sub = 0; sub < 2; ++sub) {
            int n = nc + sub * 16 + nl;
            const float* p = &xyz[((size_t)b * Nn + n) * 3];
            float px = p[0], py = p[1], pz = p[2];
            float x2 = xyz2[b * Nn + n];
            float s1v = fmaxf(s1[b * Nn + n], 1e-12f);
            #pragma unroll
            for (int v = 0; v < 8; ++v) {
                int m = (hi << 3) + v;
                float d = sC[m][3] + x2 - 2.0f * (sC[m][0] * px + sC[m][1] * py + sC[m][2] * pz);
                float maskf = (d < sC[m][4]) ? 1.0f : 0.0f;
                float l = (maskf > 0.0f) ? (sub ? lg1[v] : lg0[v]) : -1.0e9f;
                float pr = expf(l - sC[m][5]) / sC[m][6];
                float a  = sqrtf(maskf + 1e-9f) * sqrtf(pr + 1e-9f) - 1e-9f;
                float ap = a / s1v;
                sacc8[v] += ap;
                sA[m][sub * 16 + nl] = (__bf16)ap;
            }
        }
        __syncthreads();

        // --- smoothed accumulation: adj tile x x-chunk over 64 channels
        bf16x16 aA = lds_load_tile16x32(&sA[0][0], lane);
        #pragma unroll
        for (int ct = 0; ct < 4; ++ct) {
            bf16x16 bX = lds_load_tile16x32(&sXt[ct * 16][0], lane);
            acc[ct] = __builtin_amdgcn_wmma_f32_16x16x32_bf16(false, aA, false, bX,
                                                              (short)0, acc[ct], false, false);
        }
        __syncthreads();
    }

    // s2 row sums: reduce within each 16-lane n-group (hi stays constant)
    #pragma unroll
    for (int v = 0; v < 8; ++v) {
        #pragma unroll
        for (int off = 8; off; off >>= 1) sacc8[v] += __shfl_xor(sacc8[v], off, 32);
    }

    #pragma unroll
    for (int ct = 0; ct < 4; ++ct) {
        #pragma unroll
        for (int v = 0; v < 8; ++v) {
            int mrow = (hi << 3) + v;
            int c = ct * 16 + nl;
            int gm = b * Mm + mt + mrow;
            float sm = acc[ct][v] / fmaxf(sacc8[v], 1e-12f);
            float cf = x[((size_t)b * Nn + mt + mrow) * Cc + c];
            in_feat[(size_t)gm * OUTC + c]      = sm - cf;
            in_feat[(size_t)gm * OUTC + Cc + c] = cf;
        }
    }
}

// Generic bf16 WMMA GEMM with optional per-output-channel BN stats.
// C[r,o] = sum_k A[r,k]*W[o,k].  grid = (R/16, O/16), block = 32 (one wave).
// Both operand tiles staged k-contiguous (W transposed) with packed b128 stores.
__global__ __launch_bounds__(32)
void gemm_bf16_stats_kernel(const float* __restrict__ A, const float* __restrict__ W,
                            float* __restrict__ Cout, float* __restrict__ sum,
                            float* __restrict__ ssq, int R, int O, int K, int doStats) {
    int rt = blockIdx.x * 16, ot = blockIdx.y * 16, lane = threadIdx.x;
    __shared__ __align__(16) __bf16 sA[16][32];
    __shared__ __align__(16) __bf16 sBt[16][32];       // W tile, o-major (k contiguous)
    floatx8 acc = {};

    for (int kc = 0; kc < K; kc += 32) {
        #pragma unroll
        for (int i = 0; i < 2; ++i) {                  // 64 groups of 8, 2 per lane
            int g = lane + 32 * i;
            int r = g >> 2;                            // 0..15
            int k0 = (g & 3) << 3;                     // 0,8,16,24
            uint4 ua = make_uint4(0u, 0u, 0u, 0u);
            uint4 uw = make_uint4(0u, 0u, 0u, 0u);
            if (kc + k0 < K) {                         // K is a multiple of 8
                const float4* sa = (const float4*)&A[(size_t)(rt + r) * K + kc + k0];
                float4 a0 = sa[0], a1 = sa[1];
                ua = make_uint4(pack_bf16x2(a0.x, a0.y), pack_bf16x2(a0.z, a0.w),
                                pack_bf16x2(a1.x, a1.y), pack_bf16x2(a1.z, a1.w));
                const float4* sw = (const float4*)&W[(size_t)(ot + r) * K + kc + k0];
                float4 w0 = sw[0], w1 = sw[1];
                uw = make_uint4(pack_bf16x2(w0.x, w0.y), pack_bf16x2(w0.z, w0.w),
                                pack_bf16x2(w1.x, w1.y), pack_bf16x2(w1.z, w1.w));
            }
            *(uint4*)&sA[r][k0]  = ua;
            *(uint4*)&sBt[r][k0] = uw;
        }
        __syncthreads();
        bf16x16 a = lds_load_tile16x32(&sA[0][0], lane);
        bf16x16 b = lds_load_tile16x32(&sBt[0][0], lane);
        acc = __builtin_amdgcn_wmma_f32_16x16x32_bf16(
            false, a, false, b, (short)0, acc, false, false);
        __syncthreads();
    }
    int n = lane & 15, mh = (lane >> 4) << 3;
    float ps = 0.0f, pss = 0.0f;
    #pragma unroll
    for (int v = 0; v < 8; ++v) {
        float y = acc[v];
        Cout[(size_t)(rt + mh + v) * O + ot + n] = y;
        ps += y; pss += y * y;
    }
    if (doStats) {
        atomicAdd(&sum[ot + n], ps);
        atomicAdd(&ssq[ot + n], pss);
    }
}

// first-20 (ascending n) with dist <= radius; ballot-ordered scan, wave per (b,m)
__global__ __launch_bounds__(32)
void neighbor_idx_kernel(const float* __restrict__ xyz, const float* __restrict__ xyz2,
                         const float* __restrict__ radius, int* __restrict__ idxbuf) {
    int gm = blockIdx.x;
    int b = gm >> 11, m = gm & (Mm - 1), lane = threadIdx.x;
    __shared__ int sIdx[KNB];
    const float* cen = &xyz[((size_t)b * Nn + m) * 3];
    float cx = cen[0], cy = cen[1], cz = cen[2];
    float c2 = xyz2[b * Nn + m], rad = radius[gm];
    int found = 0;
    for (int nc = 0; nc < Nn && found < KNB; nc += 32) {
        int n = nc + lane;
        const float* p = &xyz[((size_t)b * Nn + n) * 3];
        float d = c2 + xyz2[b * Nn + n] - 2.0f * (cx * p[0] + cy * p[1] + cz * p[2]);
        bool pred = !(d > rad);
        unsigned bal = (unsigned)__ballot(pred);
        int before = __popc(bal & ((1u << lane) - 1u));
        if (pred && (found + before) < KNB) sIdx[found + before] = n;
        found += __popc(bal);
    }
    __syncthreads();
    if (found > KNB) found = KNB;
    if (lane < KNB) {
        int v = (lane < found) ? sIdx[lane] : sIdx[0];
        idxbuf[(size_t)gm * KNB + lane] = v;
    }
}

// h_raw[b,m,k,16] = f9 @ Wp^T + bp, plus channel stats
__global__ __launch_bounds__(256)
void hraw_kernel(const float* __restrict__ xyz, const int* __restrict__ idxbuf,
                 const float* __restrict__ Wp, const float* __restrict__ bp,
                 float* __restrict__ hraw, float* __restrict__ stats /* 32 */) {
    __shared__ float red[32];
    if (threadIdx.x < 32) red[threadIdx.x] = 0.0f;
    __syncthreads();
    int t = blockIdx.x * 256 + threadIdx.x;            // 0..8191
    int b = t >> 11, m = t & (Mm - 1);
    const float* cen = &xyz[((size_t)b * Nn + m) * 3];
    float f9[9];
    f9[0] = cen[0]; f9[1] = cen[1]; f9[2] = cen[2];
    float ls[HID], lss[HID];
    #pragma unroll
    for (int o = 0; o < HID; ++o) { ls[o] = 0.0f; lss[o] = 0.0f; }
    for (int k = 0; k < KNB; ++k) {
        int n = idxbuf[(size_t)t * KNB + k];
        const float* p = &xyz[((size_t)b * Nn + n) * 3];
        f9[3] = p[0]; f9[4] = p[1]; f9[5] = p[2];
        f9[6] = p[0] - f9[0]; f9[7] = p[1] - f9[1]; f9[8] = p[2] - f9[2];
        #pragma unroll
        for (int o = 0; o < HID; ++o) {
            float acc = bp[o];
            #pragma unroll
            for (int j = 0; j < 9; ++j) acc += Wp[o * 9 + j] * f9[j];
            hraw[((size_t)t * KNB + k) * HID + o] = acc;
            ls[o] += acc; lss[o] += acc * acc;
        }
    }
    #pragma unroll
    for (int o = 0; o < HID; ++o) {
        atomicAdd(&red[o], ls[o]);
        atomicAdd(&red[16 + o], lss[o]);
    }
    __syncthreads();
    if (threadIdx.x < 32) atomicAdd(&stats[threadIdx.x], red[threadIdx.x]);
}

__global__ __launch_bounds__(256)
void hmax_kernel(const float* __restrict__ hraw, const float* __restrict__ sc,
                 const float* __restrict__ sh, float* __restrict__ h1) {
    int t = blockIdx.x * 256 + threadIdx.x;            // 0..8191
    #pragma unroll
    for (int o = 0; o < HID; ++o) {
        float mx = -3.0e38f;
        for (int k = 0; k < KNB; ++k) {
            float v = hraw[((size_t)t * KNB + k) * HID + o] * sc[o] + sh[o];
            mx = fmaxf(mx, geluf(v));
        }
        h1[(size_t)t * HID + o] = mx;
    }
}

__global__ __launch_bounds__(256)
void final_fuse_kernel(const float* __restrict__ yfeat, const float* __restrict__ pos,
                       const float* __restrict__ fs, const float* __restrict__ fb,
                       const float* __restrict__ ps, const float* __restrict__ pb,
                       float* __restrict__ out) {
    int i = blockIdx.x * 256 + threadIdx.x;            // 0..1048575
    int o = i & (OUTC - 1);
    float v = yfeat[i] * fs[o] + fb[o] + pos[i] * ps[o] + pb[o];
    out[i] = geluf(v);
}

__global__ __launch_bounds__(256)
void centroid_copy_kernel(const float* __restrict__ xyz, float* __restrict__ out) {
    int i = blockIdx.x * 256 + threadIdx.x;            // 0..24575
    int b = i / (Mm * 3);
    int rem = i - b * (Mm * 3);
    out[i] = xyz[(size_t)b * Nn * 3 + rem];
}

// ---------------------------------------------------------------------------
extern "C" void kernel_launch(void* const* d_in, const int* in_sizes, int n_in,
                              void* d_out, int out_size, void* d_ws, size_t ws_size,
                              hipStream_t stream) {
    const float* x    = (const float*)d_in[0];
    const float* xyz  = (const float*)d_in[1];
    // d_in[2] = cent_num (hardcoded 2048)
    const float* Wq  = (const float*)d_in[3];
    const float* gq  = (const float*)d_in[4];
    const float* bq  = (const float*)d_in[5];
    const float* Wk  = (const float*)d_in[6];
    const float* gk  = (const float*)d_in[7];
    const float* bk  = (const float*)d_in[8];
    const float* Wv  = (const float*)d_in[9];
    const float* gv  = (const float*)d_in[10];
    const float* bv  = (const float*)d_in[11];
    const float* Wp  = (const float*)d_in[12];
    const float* bp  = (const float*)d_in[13];
    const float* gp  = (const float*)d_in[14];
    const float* betp= (const float*)d_in[15];
    const float* Wm1 = (const float*)d_in[16];
    const float* gm1 = (const float*)d_in[17];
    const float* bm1 = (const float*)d_in[18];
    const float* Wm2 = (const float*)d_in[19];
    const float* gm2 = (const float*)d_in[20];
    const float* bm2 = (const float*)d_in[21];
    float* out = (float*)d_out;
    float* ws  = (float*)d_ws;

    // ---- workspace layout (floats) ----
    size_t o_xyz2   = 0;                      // 16384
    size_t o_qraw   = o_xyz2 + 16384;         // 131072
    size_t o_kraw   = o_qraw + 131072;        // 262144
    size_t o_qn     = o_kraw + 262144;        // 131072
    size_t o_kn     = o_qn   + 131072;        // 262144
    size_t o_kd     = o_kn   + 262144;        // 8192
    size_t o_rad    = o_kd   + 8192;          // 8192
    size_t o_rmax   = o_rad  + 8192;          // 8192
    size_t o_rsum   = o_rmax + 8192;          // 8192
    size_t o_s1     = o_rsum + 8192;          // 16384
    size_t o_infeat = o_s1   + 16384;         // 1048576
    size_t o_yfeat  = o_infeat + 1048576;     // 1048576
    size_t o_h1     = o_yfeat  + 1048576;     // 131072
    size_t o_h2raw  = o_h1     + 131072;      // 262144
    size_t o_h2     = o_h2raw  + 262144;      // 262144
    size_t o_pos    = o_h2     + 262144;      // 1048576
    size_t o_hraw   = o_pos    + 1048576;     // 2621440
    size_t o_stats  = o_hraw   + 2621440;     // 2048
    size_t o_idx    = o_stats  + 2048;        // ints: 163840

    float* st = ws + o_stats;
    float* qsum = st + 0,   * qss = st + 16;
    float* ksum = st + 32,  * kss = st + 48;
    float* fsum = st + 64,  * fss = st + 192;
    float* hsum = st + 320, * hss = st + 336;
    float* m1sum= st + 352, * m1ss= st + 384;
    float* psum = st + 416, * pss = st + 544;
    float* qsc = st + 672,  * qsh = st + 688;
    float* ksc = st + 704,  * ksh = st + 720;
    float* fsc = st + 736,  * fsh = st + 864;
    float* hsc = st + 992,  * hsh = st + 1008;
    float* m1sc= st + 1024, * m1sh= st + 1056;
    float* psc = st + 1088, * psh = st + 1216;
    int* idxb = (int*)(ws + o_idx);

    // 0) zero stat accumulators
    zero_kernel<<<4, 256, 0, stream>>>(st, 672);
    // 1) squared norms
    xyz2_kernel<<<64, 256, 0, stream>>>(xyz, ws + o_xyz2);
    // 2) q/k raw + stats
    qk_raw_kernel<<<64, 256, 0, stream>>>(x, xyz, Wq, Wk, ws + o_qraw, ws + o_kraw, st);
    // 3) BN finalize + apply for q,k
    bn_finalize_kernel<<<1, 32, 0, stream>>>(qsum, qss, gq, bq, qsc, qsh, 8192.0f, HID);
    bn_finalize_kernel<<<1, 32, 0, stream>>>(ksum, kss, gk, bk, ksc, ksh, 16384.0f, HID);
    apply_affine_kernel<<<512, 256, 0, stream>>>(ws + o_qraw, ws + o_qn, qsc, qsh, 131072, HID, 0);
    apply_affine_kernel<<<1024, 256, 0, stream>>>(ws + o_kraw, ws + o_kn, ksc, ksh, 262144, HID, 0);
    // 4) kernel density + adaptive radius
    kd_kernel<<<Bb * Mm, 32, 0, stream>>>(xyz, ws + o_xyz2, ws + o_kd);
    radius_kernel<<<Bb, 256, 0, stream>>>(ws + o_kd, ws + o_rad);
    // 5) masked softmax row stats (online)
    softmax_stats_kernel<<<Bb * Mm, 32, 0, stream>>>(xyz, ws + o_xyz2, ws + o_qn, ws + o_kn,
                                                     ws + o_rad, ws + o_rmax, ws + o_rsum);
    // 6) column sums s1
    colsum_kernel<<<Bb * Nn, 32, 0, stream>>>(xyz, ws + o_xyz2, ws + o_qn, ws + o_kn,
                                              ws + o_rad, ws + o_rmax, ws + o_rsum, ws + o_s1);
    // 7) smoothed = adj'' @ x  (6 WMMAs / chunk) -> feat-GEMM input [sm-cf | cf]
    smoothed_wmma_kernel<<<dim3(Mm / 16, Bb), 32, 0, stream>>>(
        x, xyz, ws + o_xyz2, ws + o_qn, ws + o_kn, ws + o_rad,
        ws + o_rmax, ws + o_rsum, ws + o_s1, ws + o_infeat);
    // 8) feat GEMM (WMMA) + BN stats, finalize
    gemm_bf16_stats_kernel<<<dim3(512, 8), 32, 0, stream>>>(
        ws + o_infeat, Wv, ws + o_yfeat, fsum, fss, 8192, OUTC, OUTC, 1);
    bn_finalize_kernel<<<1, 128, 0, stream>>>(fsum, fss, gv, bv, fsc, fsh, 8192.0f, OUTC);
    // 9) positional branch: KNN indices -> h -> max -> MLP
    neighbor_idx_kernel<<<Bb * Mm, 32, 0, stream>>>(xyz, ws + o_xyz2, ws + o_rad, idxb);
    hraw_kernel<<<32, 256, 0, stream>>>(xyz, idxb, Wp, bp, ws + o_hraw, hsum);
    bn_finalize_kernel<<<1, 32, 0, stream>>>(hsum, hss, gp, betp, hsc, hsh, 163840.0f, HID);
    hmax_kernel<<<32, 256, 0, stream>>>(ws + o_hraw, hsc, hsh, ws + o_h1);
    gemm_bf16_stats_kernel<<<dim3(512, 2), 32, 0, stream>>>(
        ws + o_h1, Wm1, ws + o_h2raw, m1sum, m1ss, 8192, 32, HID, 1);
    bn_finalize_kernel<<<1, 32, 0, stream>>>(m1sum, m1ss, gm1, bm1, m1sc, m1sh, 8192.0f, 32);
    apply_affine_kernel<<<1024, 256, 0, stream>>>(ws + o_h2raw, ws + o_h2, m1sc, m1sh, 262144, 32, 1);
    gemm_bf16_stats_kernel<<<dim3(512, 8), 32, 0, stream>>>(
        ws + o_h2, Wm2, ws + o_pos, psum, pss, 8192, OUTC, 32, 1);
    bn_finalize_kernel<<<1, 128, 0, stream>>>(psum, pss, gm2, bm2, psc, psh, 8192.0f, OUTC);
    // 10) out = gelu(BN(feat) + BN(pos)); then centroid copy
    final_fuse_kernel<<<4096, 256, 0, stream>>>(ws + o_yfeat, ws + o_pos, fsc, fsh, psc, psh, out);
    centroid_copy_kernel<<<96, 256, 0, stream>>>(xyz, out + (size_t)Bb * Mm * OUTC);

    (void)in_sizes; (void)n_in; (void)out_size; (void)ws_size;
}
